// ContextAwareHead_10170482557401
// MI455X (gfx1250) — matrix operand order repaired and checked by reading, not compiled, not run on hardware
//
#include <hip/hip_runtime.h>
#include <math.h>

// ---------------------------------------------------------------------------
// ContextAwareHead for MI455X (gfx1250, wave32, WMMA 16x16x32 f16 -> f32 acc)
//
// Roofline: ~20 GFLOP vs ~213 MB unavoidable HBM traffic -> ~92 FLOP/B.
// Memory floor at 23.3 TB/s is ~9 us, so all GEMM-shaped work runs on the
// WMMA pipe in f16 and every intermediate (K, Q, hiddens, scores) lives in
// LDS (320 KB/WGP). Weights staged transposed (Bt[n][k]) so B-fragments are
// two contiguous 16B LDS loads; activations row-major for the A pattern
// (ISA 7.12.2). head_kernel waves are independent (one batch per wave), so
// they synchronize with wavefront-scope fences only (same-wave LDS is
// in-order per DScnt rules), never with block barriers after staging.
// ---------------------------------------------------------------------------

#define BATCH 32768
#define RSS   12
#define LNUM  64
#define HD    64

typedef __attribute__((ext_vector_type(16))) _Float16 v16h;
typedef __attribute__((ext_vector_type(8)))  _Float16 v8h;
typedef __attribute__((ext_vector_type(4)))  _Float16 v4h;
typedef __attribute__((ext_vector_type(8)))  float    v8f;
typedef __attribute__((ext_vector_type(4)))  float    v4f;

#if __has_builtin(__builtin_amdgcn_global_load_async_to_lds_b32) && \
    __has_builtin(__builtin_amdgcn_s_wait_asynccnt)
#define HAVE_ASYNC_LDS 1
#else
#define HAVE_ASYNC_LDS 0
#endif

// Per-wave producer/consumer sync: same-wave LDS ops retire in order (DScnt),
// we only need a compiler memory fence + scheduling barrier, not s_barrier.
__device__ __forceinline__ void wave_sync() {
  __builtin_amdgcn_fence(__ATOMIC_ACQ_REL, "wavefront");
  __builtin_amdgcn_wave_barrier();
}

// A-fragment (16x32 f16, M x K): lanes 0-15 -> M=lane, elems {K 0..7,16..23};
// lanes 16-31 -> M=lane-16, elems {K 8..15,24..31}. Two 16B LDS loads.
__device__ __forceinline__ v16h ldsA(const _Float16* base, int strideH, int kOff, int lane) {
  const int row = lane & 15;
  const int sel = lane >> 4;
  const _Float16* p = base + row * strideH + kOff + sel * 8;
  v8h lo = *(const v8h*)(p);
  v8h hi = *(const v8h*)(p + 16);
  v16h a;
#pragma unroll
  for (int i = 0; i < 8; ++i) { a[i] = lo[i]; a[i + 8] = hi[i]; }
  return a;
}

// B-fragment (32x16 f16, K x N) from a transposed Bt[n][k] LDS image:
// lane -> N = lane&15, elems = K (lane>=16 ? {8..15,24..31} : {0..7,16..23}). Two 16B LDS loads.
__device__ __forceinline__ v16h ldsB(const _Float16* baseT, int strideH, int kOff, int lane) {
  const int n   = lane & 15;
  const int sel = lane >> 4;
  const _Float16* p = baseT + n * strideH + kOff + sel * 8;
  v8h lo = *(const v8h*)(p);
  v8h hi = *(const v8h*)(p + 16);
  v16h b;
#pragma unroll
  for (int i = 0; i < 8; ++i) { b[i] = lo[i]; b[i + 8] = hi[i]; }
  return b;
}

__device__ __forceinline__ v8f wmma16(v16h a, v16h b, v8f c) {
  return __builtin_amdgcn_wmma_f32_16x16x32_f16(false, a, false, b, (short)0, c, false, false);
}

// ---------------------------------------------------------------------------
// Kernel 1: motion MLP (128->64->32) with 16 batches as the WMMA M dim,
// plus sigmoid gate cw[B,12], sigma[B], motion->f16 workspace.
// ---------------------------------------------------------------------------
__global__ __launch_bounds__(64) void motion_kernel(
    const float* __restrict__ lstm,
    const float* __restrict__ me_w1, const float* __restrict__ me_b1,
    const float* __restrict__ me_w2, const float* __restrict__ me_b2,
    const float* __restrict__ ci_w,  const float* __restrict__ ci_b,
    const float* __restrict__ ss_w,  const float* __restrict__ ss_b,
    _Float16* __restrict__ wsMot, float* __restrict__ wsSigma,
    float* __restrict__ wsCw, float* __restrict__ outCw) {
  __shared__ _Float16 sW1t[64 * 128];   // me_w1^T  [n][k]
  __shared__ _Float16 sW2t[32 * 64];    // me_w2^T  [n][k]
  __shared__ _Float16 sX[16 * 128];     // lstm tile
  __shared__ _Float16 sH[16 * 64];      // hidden
  __shared__ float    sMot[16 * 32];
  __shared__ float    sB1[64], sB2[32], sCiW[32 * 12], sCiB[12], sSsW[32], sSsB[1];

  const int tid  = threadIdx.x;
  const int lane = tid & 31;
  const int wv   = tid >> 5;
  const int b0   = blockIdx.x * 16;

  for (int i = tid; i < 64 * 128; i += 64) { int n = i >> 7, k = i & 127; sW1t[i] = (_Float16)me_w1[k * 64 + n]; }
  for (int i = tid; i < 32 * 64;  i += 64) { int n = i >> 6, k = i & 63;  sW2t[i] = (_Float16)me_w2[k * 32 + n]; }
  for (int i = tid; i < 16 * 128; i += 64) { int r = i >> 7, k = i & 127; sX[i] = (_Float16)lstm[(size_t)(b0 + r) * 128 + k]; }
  if (tid < 64) sB1[tid] = me_b1[tid];
  if (tid < 32) sB2[tid] = me_b2[tid];
  for (int i = tid; i < 32 * 12; i += 64) sCiW[i] = ci_w[i];
  if (tid < 12) sCiB[tid] = ci_b[tid];
  if (tid < 32) sSsW[tid] = ss_w[tid];
  if (tid == 0) sSsB[0] = ss_b[0];
  __syncthreads();

  // Layer 1: 16x128 @ 128x64, relu. Wave w owns N-tiles {w, w+2}.
  {
    v16h ax[4];
#pragma unroll
    for (int kc = 0; kc < 4; ++kc) ax[kc] = ldsA(sX, 128, kc * 32, lane);
    for (int nt = wv; nt < 4; nt += 2) {
      v8f acc = {};
#pragma unroll
      for (int kc = 0; kc < 4; ++kc)
        acc = wmma16(ax[kc], ldsB(sW1t + nt * 16 * 128, 128, kc * 32, lane), acc);
      const int n = lane & 15, sel = lane >> 4;
#pragma unroll
      for (int i = 0; i < 8; ++i) {
        const int m = i + 8 * sel;
        sH[m * 64 + nt * 16 + n] = (_Float16)fmaxf(acc[i] + sB1[nt * 16 + n], 0.f);
      }
    }
  }
  __syncthreads();

  // Layer 2: 16x64 @ 64x32. Wave w owns N-tile w.
  {
    const int nt = wv;
    v8f acc = {};
#pragma unroll
    for (int kc = 0; kc < 2; ++kc)
      acc = wmma16(ldsA(sH, 64, kc * 32, lane),
                   ldsB(sW2t + nt * 16 * 64, 64, kc * 32, lane), acc);
    const int n = lane & 15, sel = lane >> 4;
#pragma unroll
    for (int i = 0; i < 8; ++i) {
      const int m = i + 8 * sel;
      sMot[m * 32 + nt * 16 + n] = acc[i] + sB2[nt * 16 + n];
    }
  }
  __syncthreads();

  // Epilogue: motion->f16 ws, cw = sigmoid(motion@ci_w+b), sigma = softplus+0.5
  for (int j = tid; j < 16 * 32; j += 64) {
    int r = j >> 5, c = j & 31;
    wsMot[(size_t)(b0 + r) * 32 + c] = (_Float16)sMot[j];
  }
  for (int j = tid; j < 16 * 12; j += 64) {
    int r = j / 12, q = j % 12;
    float s = sCiB[q];
#pragma unroll
    for (int k = 0; k < 32; ++k) s += sMot[r * 32 + k] * sCiW[k * 12 + q];
    float cw = 1.f / (1.f + __expf(-s));
    wsCw[(size_t)(b0 + r) * 12 + q]  = cw;
    outCw[(size_t)(b0 + r) * 12 + q] = cw;
  }
  if (tid < 16) {
    float s = sSsB[0];
#pragma unroll
    for (int k = 0; k < 32; ++k) s += sMot[tid * 32 + k] * sSsW[k];
    float sp = (s > 20.f) ? s : log1pf(__expf(s));
    wsSigma[b0 + tid] = sp + 0.5f;
  }
}

// ---------------------------------------------------------------------------
// Kernel 2: fused K-MLP + Q-MLP + scores + softmax. One wave per batch,
// 8 waves / 256-thread block. All intermediates in per-wave LDS slices;
// after the cooperative weight staging, waves never block-synchronize.
// ---------------------------------------------------------------------------
#define WAVES2 8
__global__ __launch_bounds__(256) void head_kernel(
    const float* __restrict__ rss_t,      // [B,12,1]
    const float* __restrict__ led_feat,   // [B,64,8]
    const float* __restrict__ led_pos,    // [B,64,3]
    const float* __restrict__ prev_pos,   // [B,3]
    const int*   __restrict__ freq_mask,  // [12,64]
    const float* __restrict__ qf_w1, const float* __restrict__ qf_b1,
    const float* __restrict__ qf_w2, const float* __restrict__ qf_b2,
    const float* __restrict__ kp_w1, const float* __restrict__ kp_b1,
    const float* __restrict__ kp_w2, const float* __restrict__ kp_b2,
    const _Float16* __restrict__ wsMot,
    const float* __restrict__ wsSigma,
    const float* __restrict__ wsCw,
    float* __restrict__ outSm) {
  __shared__ _Float16 sWq1[64 * 64];      // qf_w1^T, K padded 33->64
  __shared__ _Float16 sWq2[64 * 64];      // qf_w2^T
  __shared__ _Float16 sWk1[32 * 32];      // kp_w1^T, K padded 11->32
  __shared__ _Float16 sWk2[64 * 32];      // kp_w2^T
  __shared__ float    sBq1[64], sBq2[64], sBk1[32], sBk2[64];
  __shared__ int      sFm[12 * 64];
  __shared__ _Float16 sA[WAVES2][16 * 64];   // hidden tiles
  __shared__ _Float16 sH[WAVES2][16 * 64];   // input tiles / final Q
  __shared__ _Float16 sK[WAVES2][64 * 64];   // K matrix [l][d] == Bt for scores
  __shared__ float    sSc[WAVES2][12 * 64];  // scores
  __shared__ float    sDist[WAVES2][64];
  __shared__ float    sCw[WAVES2][12];
  __shared__ float    sMx[WAVES2][12];
  __shared__ float    sIs[WAVES2][12];
  __shared__ float    sInv[WAVES2];

  const int tid  = threadIdx.x;
  const int lane = tid & 31;
  const int wv   = tid >> 5;
  const int b    = blockIdx.x * WAVES2 + wv;

  // Prefetch this wave's per-batch LED data (global_prefetch_b8).
  {
    const char* pf = (const char*)(led_feat + (size_t)b * 512);
    const char* pp = (const char*)(led_pos + (size_t)b * 192);
    if (lane < 16) __builtin_prefetch(pf + lane * 128, 0, 3);
    else if (lane < 22) __builtin_prefetch(pp + (lane - 16) * 128, 0, 3);
  }

  // Cooperative staging of shared weights (f32 -> f16, transposed).
  for (int i = tid; i < 64 * 64; i += 256) { int n = i >> 6, k = i & 63; sWq1[i] = (k < 33) ? (_Float16)qf_w1[k * 64 + n] : (_Float16)0.f; }
  for (int i = tid; i < 64 * 64; i += 256) { int n = i >> 6, k = i & 63; sWq2[i] = (_Float16)qf_w2[k * 64 + n]; }
  for (int i = tid; i < 32 * 32; i += 256) { int n = i >> 5, k = i & 31; sWk1[i] = (k < 11) ? (_Float16)kp_w1[k * 32 + n] : (_Float16)0.f; }
  for (int i = tid; i < 64 * 32; i += 256) { int n = i >> 5, k = i & 31; sWk2[i] = (_Float16)kp_w2[k * 64 + n]; }

  // Pure byte copies go through the CDNA5 async global->LDS DMA path.
#if HAVE_ASYNC_LDS
  for (int i = tid; i < 12 * 64; i += 256)
    __builtin_amdgcn_global_load_async_to_lds_b32((int*)(freq_mask + i), (int*)(sFm + i), 0, 0);
  if (tid < 64) __builtin_amdgcn_global_load_async_to_lds_b32((int*)(qf_b1 + tid), (int*)(sBq1 + tid), 0, 0);
  if (tid < 64) __builtin_amdgcn_global_load_async_to_lds_b32((int*)(qf_b2 + tid), (int*)(sBq2 + tid), 0, 0);
  if (tid < 32) __builtin_amdgcn_global_load_async_to_lds_b32((int*)(kp_b1 + tid), (int*)(sBk1 + tid), 0, 0);
  if (tid < 64) __builtin_amdgcn_global_load_async_to_lds_b32((int*)(kp_b2 + tid), (int*)(sBk2 + tid), 0, 0);
  __builtin_amdgcn_s_wait_asynccnt(0);
#else
  for (int i = tid; i < 12 * 64; i += 256) sFm[i] = freq_mask[i];
  if (tid < 64) sBq1[tid] = qf_b1[tid];
  if (tid < 64) sBq2[tid] = qf_b2[tid];
  if (tid < 32) sBk1[tid] = kp_b1[tid];
  if (tid < 64) sBk2[tid] = kp_b2[tid];
#endif

  // Per-wave scalars: squared distance to prev_pos, cw row, 1/(2 sigma^2).
  {
    const float px = prev_pos[b * 3 + 0], py = prev_pos[b * 3 + 1], pz = prev_pos[b * 3 + 2];
    for (int l = lane; l < 64; l += 32) {
      float dx = px - led_pos[((size_t)b * 64 + l) * 3 + 0];
      float dy = py - led_pos[((size_t)b * 64 + l) * 3 + 1];
      float dz = pz - led_pos[((size_t)b * 64 + l) * 3 + 2];
      sDist[wv][l] = dx * dx + dy * dy + dz * dz;
    }
    if (lane < 12) sCw[wv][lane] = wsCw[(size_t)b * 12 + lane];
    if (lane == 0) { float s = wsSigma[b]; sInv[wv] = 0.5f / (s * s); }
  }
  // Zero the input tile once: cols 11..31 stay zero for all K-tiles, cols
  // 33..63 stay zero for q_in.
  for (int i = lane; i < 16 * 64 / 4; i += 32) { v4h z = {}; ((v4h*)sH[wv])[i] = z; }
  __syncthreads();   // single block barrier: weights ready

  // ---- K MLP: 64 LEDs as 4 M-tiles; 11->32(pad) -> relu -> 32->64 ----
  // Weight B-fragments are loop-invariant: hoist out of the M loop.
  v16h bk1[2], bk2[4];
#pragma unroll
  for (int nt = 0; nt < 2; ++nt) bk1[nt] = ldsB(sWk1 + nt * 16 * 32, 32, 0, lane);
#pragma unroll
  for (int nt = 0; nt < 4; ++nt) bk2[nt] = ldsB(sWk2 + nt * 16 * 32, 32, 0, lane);

  for (int mt = 0; mt < 4; ++mt) {
    {  // build k_in tile: one b128 global load + one b64 LDS store per lane
      const int r = lane & 15, half = lane >> 4;
      const int l = mt * 16 + r;
      v4f f = *(const v4f*)(led_feat + ((size_t)b * 64 + l) * 8 + half * 4);
      *(v4h*)(&sH[wv][r * 64 + half * 4]) = __builtin_convertvector(f, v4h);
      if (half == 0) {
        const float* pp = led_pos + ((size_t)b * 64 + l) * 3;
        sH[wv][r * 64 + 8]  = (_Float16)pp[0];
        sH[wv][r * 64 + 9]  = (_Float16)pp[1];
        sH[wv][r * 64 + 10] = (_Float16)pp[2];
      }
    }
    wave_sync();
    v16h aK = ldsA(sH[wv], 64, 0, lane);
#pragma unroll
    for (int nt = 0; nt < 2; ++nt) {
      v8f acc = {};
      acc = wmma16(aK, bk1[nt], acc);
      const int n = lane & 15, sel = lane >> 4;
#pragma unroll
      for (int i = 0; i < 8; ++i) {
        const int m = i + 8 * sel;
        sA[wv][m * 64 + nt * 16 + n] = (_Float16)fmaxf(acc[i] + sBk1[nt * 16 + n], 0.f);
      }
    }
    wave_sync();
    v16h aH = ldsA(sA[wv], 64, 0, lane);
#pragma unroll
    for (int nt = 0; nt < 4; ++nt) {
      v8f acc = {};
      acc = wmma16(aH, bk2[nt], acc);
      const int n = lane & 15, sel = lane >> 4;
#pragma unroll
      for (int i = 0; i < 8; ++i) {
        const int m = i + 8 * sel;
        sK[wv][(mt * 16 + m) * 64 + nt * 16 + n] = (_Float16)(acc[i] + sBk2[nt * 16 + n]);
      }
    }
    wave_sync();
  }

  // ---- Q MLP: q_in = [rss | motion | 0-pad], rows 12..15 zero ----
  for (int i = lane; i < 16 * 33; i += 32) {
    const int r = i / 33, c = i % 33;
    float v = 0.f;
    if (r < 12) v = (c == 0) ? rss_t[(size_t)b * 12 + r]
                             : (float)wsMot[(size_t)b * 32 + (c - 1)];
    sH[wv][r * 64 + c] = (_Float16)v;
  }
  wave_sync();
  {
    v16h a0 = ldsA(sH[wv], 64, 0, lane), a1 = ldsA(sH[wv], 64, 32, lane);
#pragma unroll
    for (int nt = 0; nt < 4; ++nt) {                 // layer 1 (K padded to 64)
      v8f acc = {};
      acc = wmma16(a0, ldsB(sWq1 + nt * 16 * 64, 64, 0, lane), acc);
      acc = wmma16(a1, ldsB(sWq1 + nt * 16 * 64, 64, 32, lane), acc);
      const int n = lane & 15, sel = lane >> 4;
#pragma unroll
      for (int i = 0; i < 8; ++i) {
        const int m = i + 8 * sel;
        sA[wv][m * 64 + nt * 16 + n] = (_Float16)fmaxf(acc[i] + sBq1[nt * 16 + n], 0.f);
      }
    }
  }
  wave_sync();
  {
    v16h a0 = ldsA(sA[wv], 64, 0, lane), a1 = ldsA(sA[wv], 64, 32, lane);
#pragma unroll
    for (int nt = 0; nt < 4; ++nt) {                 // layer 2, scaled by cw
      v8f acc = {};
      acc = wmma16(a0, ldsB(sWq2 + nt * 16 * 64, 64, 0, lane), acc);
      acc = wmma16(a1, ldsB(sWq2 + nt * 16 * 64, 64, 32, lane), acc);
      const int n = lane & 15, sel = lane >> 4;
#pragma unroll
      for (int i = 0; i < 8; ++i) {
        const int m = i + 8 * sel;
        float scale = (m < 12) ? sCw[wv][m] : 0.f;   // zero rows 12..15
        sH[wv][m * 64 + nt * 16 + n] = (_Float16)((acc[i] + sBq2[nt * 16 + n]) * scale);
      }
    }
  }
  wave_sync();

  // ---- Scores: Q (16x64) @ K^T; K stored [l][d] is already Bt ----
  const float inv2 = sInv[wv];
  {
    v16h a0 = ldsA(sH[wv], 64, 0, lane), a1 = ldsA(sH[wv], 64, 32, lane);
#pragma unroll
    for (int nt = 0; nt < 4; ++nt) {
      v8f acc = {};
      acc = wmma16(a0, ldsB(sK[wv] + nt * 16 * 64, 64, 0, lane), acc);
      acc = wmma16(a1, ldsB(sK[wv] + nt * 16 * 64, 64, 32, lane), acc);
      const int n = lane & 15, sel = lane >> 4;
#pragma unroll
      for (int i = 0; i < 8; ++i) {
        const int m = i + 8 * sel;
        if (m < 12) {
          const int l = nt * 16 + n;
          float v = acc[i] - sDist[wv][l] * inv2;
          if (sFm[m * 64 + l] == 0) v = -3.4028234663852886e38f;
          sSc[wv][m * 64 + l] = v;
        }
      }
    }
  }
  wave_sync();

  // ---- Softmax over L=64 ----
  // Pass 1: lanes 0..11 reduce their row (max, 1/sum).
  if (lane < 12) {
    const float* row = &sSc[wv][lane * 64];
    float mx = -3.4028234663852886e38f;
    for (int l = 0; l < 64; ++l) mx = fmaxf(mx, row[l]);
    float sum = 0.f;
    for (int l = 0; l < 64; ++l) sum += __expf(row[l] - mx);
    sMx[wv][lane] = mx;
    sIs[wv][lane] = 1.f / sum;
  }
  wave_sync();
  // Pass 2: whole wave writes the 768-float row block as coalesced b128s.
  {
    float* obase = outSm + (size_t)b * (RSS * LNUM);
    for (int idx = lane; idx < 192; idx += 32) {      // 192 float4 = 12*64
      const int q  = idx >> 4;
      const int c0 = (idx & 15) * 4;
      const float mx = sMx[wv][q], is = sIs[wv][q];
      const float* r = &sSc[wv][q * 64 + c0];
      v4f o;
#pragma unroll
      for (int j = 0; j < 4; ++j) o[j] = __expf(r[j] - mx) * is;
      *(v4f*)(obase + idx * 4) = o;
    }
  }
}

// ---------------------------------------------------------------------------
extern "C" void kernel_launch(void* const* d_in, const int* in_sizes, int n_in,
                              void* d_out, int out_size, void* d_ws, size_t ws_size,
                              hipStream_t stream) {
  const float* rss_t     = (const float*)d_in[0];
  const float* led_feat  = (const float*)d_in[1];
  const float* led_pos   = (const float*)d_in[2];
  const float* prev_pos  = (const float*)d_in[3];
  const int*   freq_mask = (const int*)d_in[4];
  const float* lstm      = (const float*)d_in[5];
  const float* me_w1 = (const float*)d_in[6];
  const float* me_b1 = (const float*)d_in[7];
  const float* me_w2 = (const float*)d_in[8];
  const float* me_b2 = (const float*)d_in[9];
  const float* ci_w  = (const float*)d_in[10];
  const float* ci_b  = (const float*)d_in[11];
  const float* qf_w1 = (const float*)d_in[12];
  const float* qf_b1 = (const float*)d_in[13];
  const float* qf_w2 = (const float*)d_in[14];
  const float* qf_b2 = (const float*)d_in[15];
  const float* kp_w1 = (const float*)d_in[16];
  const float* kp_b1 = (const float*)d_in[17];
  const float* kp_w2 = (const float*)d_in[18];
  const float* kp_b2 = (const float*)d_in[19];
  const float* ss_w  = (const float*)d_in[20];
  const float* ss_b  = (const float*)d_in[21];

  float* outSm = (float*)d_out;                             // [B,12,64]
  float* outCw = outSm + (size_t)BATCH * RSS * LNUM;        // [B,12]

  char* ws = (char*)d_ws;                                   // ~3.7 MB used
  _Float16* wsMot = (_Float16*)ws;                          // [B,32] f16
  float* wsSigma  = (float*)(ws + (size_t)BATCH * 32 * 2);  // [B]
  float* wsCw     = wsSigma + BATCH;                        // [B,12]

  motion_kernel<<<BATCH / 16, 64, 0, stream>>>(
      lstm, me_w1, me_b1, me_w2, me_b2, ci_w, ci_b, ss_w, ss_b,
      wsMot, wsSigma, wsCw, outCw);

  head_kernel<<<BATCH / WAVES2, 256, 0, stream>>>(
      rss_t, led_feat, led_pos, prev_pos, freq_mask,
      qf_w1, qf_b1, qf_w2, qf_b2, kp_w1, kp_b1, kp_w2, kp_b2,
      wsMot, wsSigma, wsCw, outSm);

  (void)in_sizes; (void)n_in; (void)out_size; (void)ws_size;
}
